// MCMLP_29686813949991
// MI455X (gfx1250) — matrix operand order repaired and checked by reading, not compiled, not run on hardware
//
#include <hip/hip_runtime.h>
#include <hip/hip_bf16.h>

// ---------------------------------------------------------------------------
// MC-MLP forward for MI455X (gfx1250), wave32 + WMMA bf16x3 split precision.
//
//   prep:  X -> Xhi/Xlo planes;  W -> WT hi/lo planes (transposed, mask baked)
//   h0 = X @ W0 + b0                       (GEMM, K=1024)
//   h1 = relu(h0) @ (W1*mask) + b1         (K-loop bounded at tileN+128)
//   h2 = relu(h1) @ (W2*mask) + b2
//   out = relu(h2) . W3[:,4095] + b3[4095]
//
// GEMM inner loop: zero conversion VALU; staging is pure 128b copies,
// async global->LDS (double buffered) when the builtin is available.
// ---------------------------------------------------------------------------

typedef __attribute__((ext_vector_type(16))) __bf16 v16bf;
typedef __attribute__((ext_vector_type(8)))  __bf16 v8bf;
typedef __attribute__((ext_vector_type(8)))  float  v8f;
typedef int v4i_ __attribute__((vector_size(16)));

#define HID   4096
#define MROWS 8192

constexpr int BM = 128;
constexpr int BN = 128;
constexpr int BK = 32;
constexpr int LDSK = BK + 8;     // padded k-stride (elems); 80B rows, 16B aligned
constexpr int THREADS = 256;     // 8 waves (wave32)

#if defined(__HIP_DEVICE_COMPILE__) && defined(__gfx1250__) && \
    __has_builtin(__builtin_amdgcn_global_load_async_to_lds_b128) && \
    __has_builtin(__builtin_amdgcn_s_wait_asynccnt)
#define USE_ASYNC_LDS 1
#else
#define USE_ASYNC_LDS 0
#endif

// Bit-exact, codegen-safe f32 <-> bf16 (round-to-nearest-even).
__device__ __forceinline__ __bf16 f2bf(float x) {
  union { float f; unsigned u; } c; c.f = x;
  unsigned r = c.u + 0x7FFFu + ((c.u >> 16) & 1u);
  union { unsigned short s; __bf16 b; } o; o.s = (unsigned short)(r >> 16);
  return o.b;
}
__device__ __forceinline__ float bf2f(__bf16 b) {
  union { unsigned short s; __bf16 b; } i; i.b = b;
  union { unsigned u; float f; } o; o.u = ((unsigned)i.s) << 16;
  return o.f;
}

__device__ __forceinline__ void cp16(__bf16* l, const __bf16* g) {
#if USE_ASYNC_LDS
  __builtin_amdgcn_global_load_async_to_lds_b128(
      (__attribute__((address_space(1))) v4i_*)g,
      (__attribute__((address_space(3))) v4i_*)l, 0, 0);
#else
  *reinterpret_cast<uint4*>(l) = *reinterpret_cast<const uint4*>(g);
#endif
}

// ---------------- prep kernels (run once per call; memory-bound) ------------

// Elementwise split: in (f32) -> hi/lo bf16 planes, same layout.
__global__ __launch_bounds__(256)
void split_plane(const float* __restrict__ in, __bf16* __restrict__ hi,
                 __bf16* __restrict__ lo)
{
  const size_t i = ((size_t)blockIdx.x * 256 + threadIdx.x) * 4;
  const float4 v = *reinterpret_cast<const float4*>(in + i);
  const float vv[4] = {v.x, v.y, v.z, v.w};
  #pragma unroll
  for (int c = 0; c < 4; ++c) {
    const __bf16 h = f2bf(vv[c]);
    hi[i + c] = h;
    lo[i + c] = f2bf(vv[c] - bf2f(h));
  }
}

// W (K x HID, row-major) -> WT hi/lo (HID x K), block-causal mask baked in.
__global__ __launch_bounds__(256)
void split_transpose(const float* __restrict__ W, __bf16* __restrict__ Th,
                     __bf16* __restrict__ Tl, int K, int masked)
{
  __shared__ float t[32][33];
  const int n0 = blockIdx.x * 32;
  const int k0 = blockIdx.y * 32;
  const int tx = threadIdx.x;          // 0..31
  const int ty = threadIdx.y;          // 0..7
  #pragma unroll
  for (int j = 0; j < 4; ++j)
    t[ty + j * 8][tx] = W[(size_t)(k0 + ty + j * 8) * HID + n0 + tx];
  __syncthreads();
  #pragma unroll
  for (int j = 0; j < 4; ++j) {
    const int n = n0 + ty + j * 8;
    const int k = k0 + tx;
    float v = t[tx][ty + j * 8];
    if (masked && ((k >> 6) > (n >> 6))) v = 0.f;   // bake block-causal mask
    const __bf16 h = f2bf(v);
    Th[(size_t)n * K + k] = h;
    Tl[(size_t)n * K + k] = f2bf(v - bf2f(h));
  }
}

// ---------------- main GEMM: C = A@B + bias, relu, split-store --------------

__global__ __launch_bounds__(THREADS)
void gemm_bf16x3(const __bf16* __restrict__ Ahg, const __bf16* __restrict__ Alg,
                 const __bf16* __restrict__ Bhg, const __bf16* __restrict__ Blg,
                 const float* __restrict__ bias,
                 __bf16* __restrict__ Ohi, __bf16* __restrict__ Olo,
                 int K, int masked)
{
  __shared__ __attribute__((aligned(16))) __bf16 AhS[2][BM * LDSK];
  __shared__ __attribute__((aligned(16))) __bf16 AlS[2][BM * LDSK];
  __shared__ __attribute__((aligned(16))) __bf16 BhS[2][BN * LDSK];
  __shared__ __attribute__((aligned(16))) __bf16 BlS[2][BN * LDSK];

  const int tid   = threadIdx.x;
  const int lane  = tid & 31;
  const int wid   = tid >> 5;
  const int waveM = wid & 3;           // 4 waves x 32 rows
  const int waveN = wid >> 2;          // 2 waves x 64 cols

  const int tileM = blockIdx.y * BM;
  const int tileN = blockIdx.x * BN;
  const int Keff  = masked ? (tileN + BN) : K;   // mask baked into B planes

  const int lm      = lane & 15;
  const int halfSel = lane >> 4;
  const int akk     = halfSel * 8;     // A frag: interleaved 8/16 K pattern
  const int bkk     = halfSel * 8;     // B frag: interleaved 8/16 K pattern

  v8f acc[2][4];
  #pragma unroll
  for (int mi = 0; mi < 2; ++mi)
    #pragma unroll
    for (int ni = 0; ni < 4; ++ni)
      acc[mi][ni] = (v8f){0.f, 0.f, 0.f, 0.f, 0.f, 0.f, 0.f, 0.f};

  // Pure-copy staging: 512 x 16B chunks per plane, 2 chunks/thread/plane.
  auto stage = [&](int p, int kb) {
    #pragma unroll
    for (int i = 0; i < 2; ++i) {
      const int c   = tid + THREADS * i;       // 0..511
      const int row = c >> 2;                  // 4 x 16B chunks per 32-elem row
      const int q   = (c & 3) << 3;            // element offset within row
      const size_t ga = (size_t)(tileM + row) * K + kb + q;
      const size_t gb = (size_t)(tileN + row) * K + kb + q;
      const int    lo = row * LDSK + q;
      cp16(&AhS[p][lo], Ahg + ga);
      cp16(&AlS[p][lo], Alg + ga);
      cp16(&BhS[p][lo], Bhg + gb);
      cp16(&BlS[p][lo], Blg + gb);
    }
  };

  int p = 0;
  stage(0, 0);

  for (int kb = 0; kb < Keff; kb += BK) {
#if USE_ASYNC_LDS
    __builtin_amdgcn_s_wait_asynccnt(0);
#endif
    __syncthreads();
    if (kb + BK < Keff) stage(p ^ 1, kb + BK);   // overlap copy with WMMA

    const __bf16* AhL = AhS[p];
    const __bf16* AlL = AlS[p];
    const __bf16* BhL = BhS[p];
    const __bf16* BlL = BlS[p];

    v16bf afh[2], afl[2];
    #pragma unroll
    for (int mi = 0; mi < 2; ++mi) {
      const int m = waveM * 32 + mi * 16 + lm;
      const v8bf h0 = *reinterpret_cast<const v8bf*>(&AhL[m * LDSK + akk]);
      const v8bf h1 = *reinterpret_cast<const v8bf*>(&AhL[m * LDSK + akk + 16]);
      afh[mi] = __builtin_shufflevector(h0, h1, 0, 1, 2, 3, 4, 5, 6, 7,
                                        8, 9, 10, 11, 12, 13, 14, 15);
      const v8bf l0 = *reinterpret_cast<const v8bf*>(&AlL[m * LDSK + akk]);
      const v8bf l1 = *reinterpret_cast<const v8bf*>(&AlL[m * LDSK + akk + 16]);
      afl[mi] = __builtin_shufflevector(l0, l1, 0, 1, 2, 3, 4, 5, 6, 7,
                                        8, 9, 10, 11, 12, 13, 14, 15);
    }

    #pragma unroll
    for (int ni = 0; ni < 4; ++ni) {
      const int n = waveN * 64 + ni * 16 + lm;
      const v8bf h0 = *reinterpret_cast<const v8bf*>(&BhL[n * LDSK + bkk]);
      const v8bf h1 = *reinterpret_cast<const v8bf*>(&BhL[n * LDSK + bkk + 16]);
      const v16bf bfh = __builtin_shufflevector(h0, h1, 0, 1, 2, 3, 4, 5, 6, 7,
                                                8, 9, 10, 11, 12, 13, 14, 15);
      const v8bf l0 = *reinterpret_cast<const v8bf*>(&BlL[n * LDSK + bkk]);
      const v8bf l1 = *reinterpret_cast<const v8bf*>(&BlL[n * LDSK + bkk + 16]);
      const v16bf bfl = __builtin_shufflevector(l0, l1, 0, 1, 2, 3, 4, 5, 6, 7,
                                                8, 9, 10, 11, 12, 13, 14, 15);
      #pragma unroll
      for (int mi = 0; mi < 2; ++mi) {
        acc[mi][ni] = __builtin_amdgcn_wmma_f32_16x16x32_bf16(
            false, afh[mi], false, bfh, (short)0, acc[mi][ni], false, false);
        acc[mi][ni] = __builtin_amdgcn_wmma_f32_16x16x32_bf16(
            false, afh[mi], false, bfl, (short)0, acc[mi][ni], false, false);
        acc[mi][ni] = __builtin_amdgcn_wmma_f32_16x16x32_bf16(
            false, afl[mi], false, bfh, (short)0, acc[mi][ni], false, false);
      }
    }
    p ^= 1;
  }

  // Epilogue: bias + relu, store split hi/lo activation planes.
  #pragma unroll
  for (int mi = 0; mi < 2; ++mi) {
    #pragma unroll
    for (int ni = 0; ni < 4; ++ni) {
      const int col = tileN + waveN * 64 + ni * 16 + lm;
      const float bv = bias[col];
      #pragma unroll
      for (int r = 0; r < 8; ++r) {
        const int row = tileM + waveM * 32 + mi * 16 + halfSel * 8 + r;
        float v = acc[mi][ni][r] + bv;
        v = v > 0.f ? v : 0.f;
        const __bf16 h = f2bf(v);
        Ohi[(size_t)row * HID + col] = h;
        Olo[(size_t)row * HID + col] = f2bf(v - bf2f(h));
      }
    }
  }
}

// out[r] = (hi[r,:]+lo[r,:]) . W3[:,4095] + b3[4095]
__global__ __launch_bounds__(256)
void final_dot(const __bf16* __restrict__ Ahi, const __bf16* __restrict__ Alo,
               const float* __restrict__ W3, const float* __restrict__ b3,
               float* __restrict__ out)
{
  __shared__ float red[256];
  const int r = blockIdx.x;
  float s = 0.f;
  for (int k = threadIdx.x; k < HID; k += 256) {
    const float a = bf2f(Ahi[(size_t)r * HID + k]) + bf2f(Alo[(size_t)r * HID + k]);
    s += a * W3[(size_t)k * HID + (HID - 1)];
  }
  red[threadIdx.x] = s;
  __syncthreads();
  #pragma unroll
  for (int off = 128; off > 0; off >>= 1) {
    if ((int)threadIdx.x < off) red[threadIdx.x] += red[threadIdx.x + off];
    __syncthreads();
  }
  if (threadIdx.x == 0) out[r] = red[0] + b3[HID - 1];
}

extern "C" void kernel_launch(void* const* d_in, const int* in_sizes, int n_in,
                              void* d_out, int out_size, void* d_ws, size_t ws_size,
                              hipStream_t stream) {
  const float* X  = (const float*)d_in[0];
  const float* W0 = (const float*)d_in[1];
  const float* b0 = (const float*)d_in[2];
  const float* W1 = (const float*)d_in[3];
  const float* b1 = (const float*)d_in[4];
  const float* W2 = (const float*)d_in[5];
  const float* b2 = (const float*)d_in[6];
  const float* W3 = (const float*)d_in[7];
  const float* b3 = (const float*)d_in[8];
  float* out = (float*)d_out;

  const size_t ACT = (size_t)MROWS * HID;     // activation plane elements
  const size_t WSM = (size_t)HID * 1024;      // W0 plane elements
  const size_t WBG = (size_t)HID * HID;       // W1/W2 plane elements

  __bf16* a0h = (__bf16*)d_ws;
  __bf16* a0l = a0h + ACT;
  __bf16* a1h = a0l + ACT;
  __bf16* a1l = a1h + ACT;
  __bf16* w0h = a1l + ACT;
  __bf16* w0l = w0h + WSM;
  __bf16* w1h = w0l + WSM;
  __bf16* w1l = w1h + WBG;
  __bf16* w2h = w1l + WBG;
  __bf16* w2l = w2h + WBG;
  // X split planes overlay activation set 1 (dead before stage 1 writes it).
  __bf16* xh = a1h;
  __bf16* xl = a1l;

  // ---- prep: split X; split+transpose(+mask) weights ----
  split_plane<<<(MROWS * 1024) / 4 / 256, 256, 0, stream>>>(X, xh, xl);
  split_transpose<<<dim3(HID / 32, 1024 / 32), dim3(32, 8), 0, stream>>>(
      W0, w0h, w0l, 1024, 0);
  split_transpose<<<dim3(HID / 32, HID / 32), dim3(32, 8), 0, stream>>>(
      W1, w1h, w1l, HID, 1);
  split_transpose<<<dim3(HID / 32, HID / 32), dim3(32, 8), 0, stream>>>(
      W2, w2h, w2l, HID, 1);

  // ---- 3 GEMM layers + final dot ----
  dim3 grid(HID / BN, MROWS / BM);            // (32, 64)
  gemm_bf16x3<<<grid, THREADS, 0, stream>>>(xh,  xl,  w0h, w0l, b0, a0h, a0l, 1024, 0);
  gemm_bf16x3<<<grid, THREADS, 0, stream>>>(a0h, a0l, w1h, w1l, b1, a1h, a1l, HID, 1);
  gemm_bf16x3<<<grid, THREADS, 0, stream>>>(a1h, a1l, w2h, w2l, b2, a0h, a0l, HID, 1);
  final_dot<<<MROWS, 256, 0, stream>>>(a0h, a0l, W3, b3, out);
}
